// SMHA_35493609734819
// MI455X (gfx1250) — matrix operand order repaired and checked by reading, compiled, not run on hardware
//
#include <hip/hip_runtime.h>
#include <hip/hip_bf16.h>
#include <math.h>

// ---------- problem dims ----------
#define T_DIM 2048
#define B_DIM 2
#define E_DIM 1024
#define H_DIM 16
#define D_DIM 64
#define M_DIM (T_DIM * B_DIM)   // 4096 rows for the projection GEMMs

// ---------- CDNA5 WMMA types ----------
typedef __attribute__((ext_vector_type(16))) __bf16         v16bf;
typedef __attribute__((ext_vector_type(8)))  float          v8f;
typedef __attribute__((ext_vector_type(16))) unsigned short us16;
typedef __attribute__((ext_vector_type(8)))  unsigned short us8;
typedef __attribute__((ext_vector_type(4)))  int            v4i;

#define WMMA_BF16(A, B, C) \
  __builtin_amdgcn_wmma_f32_16x16x32_bf16(false, (A), false, (B), (short)0, (C), false, false)

#if __has_builtin(__builtin_amdgcn_global_load_async_to_lds_b128)
#define ASYNC_LDS 1
typedef __attribute__((address_space(1))) v4i* as1_v4i;   // global (prints as "__device__")
typedef __attribute__((address_space(3))) v4i* as3_v4i;   // LDS    (prints as "__shared__")
__device__ __forceinline__ void async_cp16(const void* g, void* l) {
  __builtin_amdgcn_global_load_async_to_lds_b128((as1_v4i)g, (as3_v4i)l, 0, 0);
}
__device__ __forceinline__ void wait_async_all() {
#if __has_builtin(__builtin_amdgcn_s_wait_asynccnt)
  __builtin_amdgcn_s_wait_asynccnt(0);
#else
  asm volatile("s_wait_asynccnt 0x0" ::: "memory");
#endif
}
#endif

// fp32 -> bf16 (round to nearest even)
__device__ __forceinline__ unsigned short f2bf(float f) {
  unsigned int u = __builtin_bit_cast(unsigned int, f);
  u += 0x7FFFu + ((u >> 16) & 1u);
  return (unsigned short)(u >> 16);
}

// A-fragment (16x32 bf16, MxK): lane row m = lane&15; elements K = {hi*8..+7, hi*8+16..+23}
__device__ __forceinline__ v16bf a_frag(const unsigned short* rowp, int hi) {
  const us8* p0 = (const us8*)(rowp + hi * 8);
  const us8* p1 = (const us8*)(rowp + hi * 8 + 16);
  us8 a = *p0, b = *p1;
  us16 r;
#pragma unroll
  for (int i = 0; i < 8; ++i) { r[i] = a[i]; r[i + 8] = b[i]; }
  return __builtin_bit_cast(v16bf, r);
}

// B-fragment (32x16 bf16, KxN): lane col n = lane&15; elements K = {hi*8..+7, hi*8+16..+23}
__device__ __forceinline__ v16bf b_frag(const unsigned short* colp, int hi) {
  const us8* p0 = (const us8*)(colp + hi * 8);
  const us8* p1 = (const us8*)(colp + hi * 8 + 16);
  us8 a = *p0, b = *p1;
  us16 r;
#pragma unroll
  for (int i = 0; i < 8; ++i) { r[i] = a[i]; r[i + 8] = b[i]; }
  return __builtin_bit_cast(v16bf, r);
}

// ---------- fp32 -> bf16 convert ----------
__global__ void cvt_k(const float* __restrict__ in, unsigned short* __restrict__ out, int n) {
  int i = blockIdx.x * blockDim.x + threadIdx.x;
  if (i < n) out[i] = f2bf(in[i]);
}

// ---------- tiled WMMA GEMM:  C[M,N] = A[M,K] * W[N,K]^T (+bias) ----------
// MODE 0: Q -> bf16 [B,H,T,D], scaled by D^-0.5
// MODE 1: K -> bf16 [B,H,T,D]
// MODE 2: V -> bf16 [B,H,D,T]  (transposed for contiguous P*V B-fragments)
// MODE 3: f32 output [M,E] = d_out
template <int MODE>
__global__ __launch_bounds__(256) void gemm_k(const unsigned short* __restrict__ A,
                                              const unsigned short* __restrict__ W,
                                              const float* __restrict__ bias,
                                              unsigned short* __restrict__ outBF,
                                              float* __restrict__ outF) {
  __shared__ unsigned short As[2][128][40];  // double-buffered 128x32 bf16 tiles (+8 pad)
  __shared__ unsigned short Bs[2][128][40];

  const int tid  = threadIdx.x;
  const int lane = tid & 31;
  const int ln   = lane & 15;
  const int hi   = lane >> 4;
  const int wid  = tid >> 5;   // 8 waves
  const int wm   = wid >> 1;   // 0..3 -> 32 rows each
  const int wn   = wid & 1;    // 0..1 -> 64 cols each
  const int m0   = blockIdx.y * 128;
  const int n0   = blockIdx.x * 128;

  const int lr = tid >> 1;          // 0..127 tile row for cooperative loads
  const int lc = (tid & 1) * 16;    // 0 or 16

  const unsigned short* gA = A + (size_t)(m0 + lr) * E_DIM + lc;
  const unsigned short* gB = W + (size_t)(n0 + lr) * E_DIM + lc;

  v8f acc[2][4] = {};
  const int NT = E_DIM / 32;        // 32 K-steps

#if defined(ASYNC_LDS)
  // prefetch tile 0 into buffer 0 (async DMA, no VGPR round-trip)
  async_cp16(gA,     &As[0][lr][lc]);
  async_cp16(gA + 8, &As[0][lr][lc + 8]);
  async_cp16(gB,     &Bs[0][lr][lc]);
  async_cp16(gB + 8, &Bs[0][lr][lc + 8]);

  for (int kt = 0; kt < NT; ++kt) {
    const int cur = kt & 1;
    wait_async_all();      // own transfers done
    __syncthreads();       // everyone's transfers done; prev buffer fully consumed

    if (kt + 1 < NT) {     // overlap next tile's DMA with this tile's WMMAs
      const int k1 = (kt + 1) * 32;
      async_cp16(gA + k1,     &As[cur ^ 1][lr][lc]);
      async_cp16(gA + k1 + 8, &As[cur ^ 1][lr][lc + 8]);
      async_cp16(gB + k1,     &Bs[cur ^ 1][lr][lc]);
      async_cp16(gB + k1 + 8, &Bs[cur ^ 1][lr][lc + 8]);
    }

    v16bf af[2], bf_[4];
#pragma unroll
    for (int i = 0; i < 2; ++i) af[i] = a_frag(&As[cur][wm * 32 + i * 16 + ln][0], hi);
#pragma unroll
    for (int j = 0; j < 4; ++j) bf_[j] = b_frag(&Bs[cur][wn * 64 + j * 16 + ln][0], hi);
#pragma unroll
    for (int i = 0; i < 2; ++i)
#pragma unroll
      for (int j = 0; j < 4; ++j) acc[i][j] = WMMA_BF16(af[i], bf_[j], acc[i][j]);
  }
#else
  for (int kt = 0; kt < NT; ++kt) {
    const int k0 = kt * 32;
    const uint4* ga = (const uint4*)(gA + k0);
    const uint4* gb = (const uint4*)(gB + k0);
    uint4 av0 = ga[0], av1 = ga[1];
    uint4 bv0 = gb[0], bv1 = gb[1];
    *(uint4*)&As[0][lr][lc]     = av0;
    *(uint4*)&As[0][lr][lc + 8] = av1;
    *(uint4*)&Bs[0][lr][lc]     = bv0;
    *(uint4*)&Bs[0][lr][lc + 8] = bv1;
    __syncthreads();

    v16bf af[2], bf_[4];
#pragma unroll
    for (int i = 0; i < 2; ++i) af[i] = a_frag(&As[0][wm * 32 + i * 16 + ln][0], hi);
#pragma unroll
    for (int j = 0; j < 4; ++j) bf_[j] = b_frag(&Bs[0][wn * 64 + j * 16 + ln][0], hi);
#pragma unroll
    for (int i = 0; i < 2; ++i)
#pragma unroll
      for (int j = 0; j < 4; ++j) acc[i][j] = WMMA_BF16(af[i], bf_[j], acc[i][j]);
    __syncthreads();
  }
#endif

  // epilogue: C layout n = lane&15, m = r + 8*hi
#pragma unroll
  for (int i = 0; i < 2; ++i) {
#pragma unroll
    for (int j = 0; j < 4; ++j) {
      const int gn = n0 + wn * 64 + j * 16 + ln;
      const float bb = bias[gn];
#pragma unroll
      for (int r = 0; r < 8; ++r) {
        const int gm = m0 + wm * 32 + i * 16 + r + 8 * hi;
        float v = acc[i][j][r] + bb;
        if (MODE == 3) {
          outF[(size_t)gm * E_DIM + gn] = v;
        } else {
          const int t = gm >> 1, b = gm & 1;       // row = t*B + b, B==2
          const int h = gn >> 6, d = gn & 63;
          if (MODE == 0)
            outBF[((size_t)(b * H_DIM + h) * T_DIM + t) * D_DIM + d] = f2bf(v * 0.125f);
          else if (MODE == 1)
            outBF[((size_t)(b * H_DIM + h) * T_DIM + t) * D_DIM + d] = f2bf(v);
          else
            outBF[((size_t)(b * H_DIM + h) * D_DIM + d) * T_DIM + t] = f2bf(v);
        }
      }
    }
  }
}

// ---------- flash attention: per (b,h), 128 queries per block, 16 per wave ----------
__global__ __launch_bounds__(256) void attn_k(const unsigned short* __restrict__ Q,   // [BH,T,D]
                                              const unsigned short* __restrict__ K,   // [BH,T,D]
                                              const unsigned short* __restrict__ V,   // [BH,D,T]
                                              unsigned short* __restrict__ O) {       // [T*B, E] bf16
  __shared__ unsigned short Ps[8][16][40];   // per-wave P staging (C-layout -> A-layout)

  const int tid  = threadIdx.x;
  const int lane = tid & 31;
  const int ln   = lane & 15;
  const int hi   = lane >> 4;
  const int wid  = tid >> 5;
  const int bh   = blockIdx.y;            // b*H + h
  const int b    = bh / H_DIM;
  const int h    = bh % H_DIM;
  const int tw   = blockIdx.x * 128 + wid * 16;

  // Q A-fragments (rows = queries, K = head dim, 2 chunks of 32)
  const unsigned short* Qp = Q + ((size_t)bh * T_DIM + tw + ln) * D_DIM;
  v16bf aq[2];
  aq[0] = a_frag(Qp, hi);
  aq[1] = a_frag(Qp + 32, hi);

  v8f oacc[4] = {};
  float mrow[8], lrow[8];
#pragma unroll
  for (int r = 0; r < 8; ++r) { mrow[r] = -1.0e30f; lrow[r] = 0.0f; }

  for (int s0 = 0; s0 < T_DIM; s0 += 32) {
    // S = Q * K^T  (two 16-key tiles)
    v8f s[2] = {};
#pragma unroll
    for (int nt = 0; nt < 2; ++nt) {
      const unsigned short* Kp = K + ((size_t)bh * T_DIM + s0 + nt * 16 + ln) * D_DIM;
#pragma unroll
      for (int c = 0; c < 2; ++c) s[nt] = WMMA_BF16(aq[c], b_frag(Kp + c * 32, hi), s[nt]);
    }

    // online softmax; C layout rows m = r + 8*hi, n across the 16-lane group
    float p0a[8], p1a[8];
#pragma unroll
    for (int r = 0; r < 8; ++r) {
      float v = fmaxf(s[0][r], s[1][r]);
      v = fmaxf(v, __shfl_xor(v, 1, 32));
      v = fmaxf(v, __shfl_xor(v, 2, 32));
      v = fmaxf(v, __shfl_xor(v, 4, 32));
      v = fmaxf(v, __shfl_xor(v, 8, 32));
      const float mn = fmaxf(mrow[r], v);
      const float sc = __expf(mrow[r] - mn);
      mrow[r] = mn;
      const float p0 = __expf(s[0][r] - mn);
      const float p1 = __expf(s[1][r] - mn);
      float rs = p0 + p1;
      rs += __shfl_xor(rs, 1, 32);
      rs += __shfl_xor(rs, 2, 32);
      rs += __shfl_xor(rs, 4, 32);
      rs += __shfl_xor(rs, 8, 32);
      lrow[r] = lrow[r] * sc + rs;
      p0a[r] = p0; p1a[r] = p1;
#pragma unroll
      for (int j = 0; j < 4; ++j) oacc[j][r] *= sc;
    }

    // stage P (bf16) through wave-private LDS: C-layout stores -> A-layout loads
#pragma unroll
    for (int r = 0; r < 8; ++r) {
      Ps[wid][r + 8 * hi][ln]      = f2bf(p0a[r]);
      Ps[wid][r + 8 * hi][16 + ln] = f2bf(p1a[r]);
    }
    asm volatile("s_wait_dscnt 0x0" ::: "memory");   // intra-wave LDS RAW fence (CDNA5 split counter)
    v16bf pf = a_frag(&Ps[wid][ln][0], hi);

    // O += P * V   (V transposed: contiguous along keys)
#pragma unroll
    for (int j = 0; j < 4; ++j) {
      const unsigned short* Vp = V + ((size_t)bh * D_DIM + j * 16 + ln) * T_DIM + s0;
      oacc[j] = WMMA_BF16(pf, b_frag(Vp, hi), oacc[j]);
    }
  }

  // normalize + store bf16 attn output [T*B, E]
#pragma unroll
  for (int r = 0; r < 8; ++r) {
    const float inv = 1.0f / lrow[r];
    const int t = tw + r + 8 * hi;
#pragma unroll
    for (int j = 0; j < 4; ++j)
      O[((size_t)t * B_DIM + b) * E_DIM + h * 64 + j * 16 + ln] = f2bf(oacc[j][r] * inv);
  }
}

// ---------- host launcher ----------
extern "C" void kernel_launch(void* const* d_in, const int* in_sizes, int n_in,
                              void* d_out, int out_size, void* d_ws, size_t ws_size,
                              hipStream_t stream) {
  const float* query = (const float*)d_in[0];
  const float* Wq = (const float*)d_in[1];
  const float* bq = (const float*)d_in[2];
  const float* Wk = (const float*)d_in[3];
  const float* bk = (const float*)d_in[4];
  const float* Wv = (const float*)d_in[5];
  const float* bv = (const float*)d_in[6];
  const float* Wo = (const float*)d_in[7];
  const float* bo = (const float*)d_in[8];
  float* out = (float*)d_out;

  char* ws = (char*)d_ws;
  const size_t MB = 1ull << 20;
  unsigned short* Xb  = (unsigned short*)(ws + 0);        // [4096,1024] bf16 query
  unsigned short* Wqb = (unsigned short*)(ws + 8 * MB);
  unsigned short* Wkb = (unsigned short*)(ws + 10 * MB);
  unsigned short* Wvb = (unsigned short*)(ws + 12 * MB);
  unsigned short* Wob = (unsigned short*)(ws + 14 * MB);
  unsigned short* Qh  = (unsigned short*)(ws + 16 * MB);  // [B,H,T,D]
  unsigned short* Kh  = (unsigned short*)(ws + 24 * MB);  // [B,H,T,D]
  unsigned short* Vt  = (unsigned short*)(ws + 32 * MB);  // [B,H,D,T]
  unsigned short* At  = (unsigned short*)(ws + 40 * MB);  // [4096,1024] bf16 attn out

  const int nX = M_DIM * E_DIM;        // 4M
  const int nW = E_DIM * E_DIM;        // 1M
  cvt_k<<<(nX + 255) / 256, 256, 0, stream>>>(query, Xb, nX);
  cvt_k<<<(nW + 255) / 256, 256, 0, stream>>>(Wq, Wqb, nW);
  cvt_k<<<(nW + 255) / 256, 256, 0, stream>>>(Wk, Wkb, nW);
  cvt_k<<<(nW + 255) / 256, 256, 0, stream>>>(Wv, Wvb, nW);
  cvt_k<<<(nW + 255) / 256, 256, 0, stream>>>(Wo, Wob, nW);

  dim3 gg(E_DIM / 128, M_DIM / 128);   // 8 x 32
  dim3 blk(256);
  gemm_k<0><<<gg, blk, 0, stream>>>(Xb, Wqb, bq, Qh, nullptr);
  gemm_k<1><<<gg, blk, 0, stream>>>(Xb, Wkb, bk, Kh, nullptr);
  gemm_k<2><<<gg, blk, 0, stream>>>(Xb, Wvb, bv, Vt, nullptr);

  attn_k<<<dim3(T_DIM / 128, B_DIM * H_DIM), blk, 0, stream>>>(Qh, Kh, Vt, At);

  gemm_k<3><<<gg, blk, 0, stream>>>(At, Wob, bo, nullptr, out);
}
